// ContrastiveLoss_74191265071557
// MI455X (gfx1250) — matrix operand order repaired and checked by reading, not compiled, not run on hardware
//
#include <hip/hip_runtime.h>
#include <math.h>

// ---------------------------------------------------------------------------
// Problem constants (reference: N=M=8192, D=128, MARGIN=1.0, fp32 in/out)
// ---------------------------------------------------------------------------
#define NROWS 8192
#define DDIM  128
#define MARGIN 1.0

typedef __attribute__((ext_vector_type(16))) __bf16 v16bf;
typedef __attribute__((ext_vector_type(8)))  float  v8f;

// ---- ws layout (bytes) ----------------------------------------------------
#define WS_AHI   ((size_t)0)
#define WS_ALO   ((size_t)(2u << 20))
#define WS_BHI   ((size_t)(4u << 20))
#define WS_BLO   ((size_t)(6u << 20))
#define WS_NA    ((size_t)(8u << 20))
#define WS_NB    (WS_NA + (size_t)NROWS * 4)
#define WS_POSR  (WS_NB + (size_t)NROWS * 4)
#define WS_POSC  (WS_POSR + (size_t)NROWS * 4)
#define WS_BSUM  (WS_POSC + (size_t)NROWS * 4)   // 4096 floats

#define INF_BITS 0x7F800000u

__device__ __forceinline__ unsigned short f32_to_bf16_rne(float f) {
    unsigned int u = __float_as_uint(f);
    unsigned int r = u + 0x7FFFu + ((u >> 16) & 1u);
    return (unsigned short)(r >> 16);
}
__device__ __forceinline__ float bf16_to_f32(unsigned short h) {
    return __uint_as_float(((unsigned int)h) << 16);
}

// ---------------------------------------------------------------------------
// Kernel 0: init row/col min buffers to +inf bit pattern
// ---------------------------------------------------------------------------
__global__ void init_mins_kernel(unsigned int* posR, unsigned int* posC) {
    int i = blockIdx.x * blockDim.x + threadIdx.x;
    if (i < NROWS)               posR[i] = INF_BITS;
    else if (i < 2 * NROWS)      posC[i - NROWS] = INF_BITS;
}

// ---------------------------------------------------------------------------
// Kernel 1: bf16 hi/lo split + per-row squared norms.
// grid = (8192, 2), block = 128 (one row per block; y selects matrix)
// ---------------------------------------------------------------------------
__global__ void prep_kernel(const float* __restrict__ A, const float* __restrict__ B,
                            unsigned short* __restrict__ Ahi, unsigned short* __restrict__ Alo,
                            unsigned short* __restrict__ Bhi, unsigned short* __restrict__ Blo,
                            float* __restrict__ normA, float* __restrict__ normB) {
    const int row = blockIdx.x;
    const int mat = blockIdx.y;
    const int t   = threadIdx.x;
    const float* src       = mat ? B   : A;
    unsigned short* dstHi  = mat ? Bhi : Ahi;
    unsigned short* dstLo  = mat ? Blo : Alo;
    float* dstN            = mat ? normB : normA;

    const size_t idx = (size_t)row * DDIM + t;
    float x = src[idx];
    unsigned short hi = f32_to_bf16_rne(x);
    float resid = x - bf16_to_f32(hi);
    unsigned short lo = f32_to_bf16_rne(resid);
    dstHi[idx] = hi;
    dstLo[idx] = lo;

    __shared__ float sh[DDIM];
    sh[t] = x * x;
    __syncthreads();
    #pragma unroll
    for (int s = DDIM / 2; s > 0; s >>= 1) {
        if (t < s) sh[t] += sh[t + s];
        __syncthreads();
    }
    if (t == 0) dstN[row] = sh[0];
}

// ---------------------------------------------------------------------------
// Fragment loader: two b128 loads at compile-time element offsets from a base
// that already folds in the per-lane half term (16h for A, 16h for B).
// A 16x32 bf16: lane half h holds K in {16h .. 16h+15} (contiguous)
// B 32x16 bf16: lane half h holds K in {16h .. 16h+15} (contiguous)
// ---------------------------------------------------------------------------
__device__ __forceinline__ v16bf load_frag(const unsigned short* __restrict__ base,
                                           int eoff0, int eoff1) {
    union { uint4 q[2]; v16bf v; } u;
    u.q[0] = *(const uint4*)(base + eoff0);
    u.q[1] = *(const uint4*)(base + eoff1);
    return u.v;
}

// ---------------------------------------------------------------------------
// Kernel 2: 128x128 distance tile. 256 threads = 8 waves; wave (wr,wc) owns a
// 32x64 strip = 2x4 fragments of 16x16. Split-bf16 GEMM:
//   dot ~= hiA*hiB + hiA*loB + loA*hiB   (f32 WMMA accumulators)
// Epilogue: d = sqrt(max(|a|^2+|b|^2-2dot,0)); reduce sum/rowmin/colmin.
// __launch_bounds__(256,1): allow a large register file, no spills; this
// kernel is WMMA-bound with L2-resident operands, occupancy is secondary.
// ---------------------------------------------------------------------------
__global__ void __launch_bounds__(256, 1)
dist_tile_kernel(const unsigned short* __restrict__ Ahi,
                 const unsigned short* __restrict__ Alo,
                 const unsigned short* __restrict__ Bhi,
                 const unsigned short* __restrict__ Blo,
                 const float* __restrict__ normA,
                 const float* __restrict__ normB,
                 unsigned int* __restrict__ posR,
                 unsigned int* __restrict__ posC,
                 float* __restrict__ blockSums) {
    __shared__ unsigned int lds_min_r[128];
    __shared__ unsigned int lds_min_c[128];
    __shared__ float ssum[256];

    const int tid  = threadIdx.x;
    const int lane = tid & 31;
    const int wid  = tid >> 5;
    const int wr   = wid >> 1;      // 0..3
    const int wc   = wid & 1;       // 0..1
    const int h    = lane >> 4;     // 0..1
    const int l15  = lane & 15;

    const int tileM = blockIdx.y * 128;
    const int tileN = blockIdx.x * 128;

    if (tid < 128) { lds_min_r[tid] = INF_BITS; lds_min_c[tid] = INF_BITS; }
    __syncthreads();

    // Per-fragment base pointers (computed once; kc stride is an immediate).
    const unsigned short* aHiB[2];
    const unsigned short* aLoB[2];
    #pragma unroll
    for (int fr = 0; fr < 2; ++fr) {
        const unsigned int ro =
            (unsigned int)(tileM + 32 * wr + 16 * fr + l15) * DDIM + 16u * h;
        aHiB[fr] = Ahi + ro;
        aLoB[fr] = Alo + ro;
    }
    const unsigned short* bHiB[4];
    const unsigned short* bLoB[4];
    #pragma unroll
    for (int fc = 0; fc < 4; ++fc) {
        const unsigned int co =
            (unsigned int)(tileN + 64 * wc + 16 * fc + l15) * DDIM + 16u * h;
        bHiB[fc] = Bhi + co;
        bLoB[fc] = Blo + co;
    }

    const v8f vzero = {0.f, 0.f, 0.f, 0.f, 0.f, 0.f, 0.f, 0.f};
    v8f acc[2][4];
    #pragma unroll
    for (int fr = 0; fr < 2; ++fr)
        #pragma unroll
        for (int fc = 0; fc < 4; ++fc) acc[fr][fc] = vzero;

    #pragma unroll
    for (int kc = 0; kc < DDIM / 32; ++kc) {
        const int ka0 = kc * 32;        // A: first 8 elems at base+ka0
        const int ka1 = kc * 32 + 8;    // A: second 8 elems
        const int kb0 = kc * 32;        // B: 16 contiguous elems
        const int kb1 = kc * 32 + 8;

        v16bf ahi[2], alo[2];
        #pragma unroll
        for (int fr = 0; fr < 2; ++fr) {
            ahi[fr] = load_frag(aHiB[fr], ka0, ka1);
            alo[fr] = load_frag(aLoB[fr], ka0, ka1);
        }
        #pragma unroll
        for (int fc = 0; fc < 4; ++fc) {
            v16bf bhi = load_frag(bHiB[fc], kb0, kb1);
            v16bf blo = load_frag(bLoB[fc], kb0, kb1);
            #pragma unroll
            for (int fr = 0; fr < 2; ++fr) {
                acc[fr][fc] = __builtin_amdgcn_wmma_f32_16x16x32_bf16(
                    false, ahi[fr], false, bhi, (short)0, acc[fr][fc], false, false);
                acc[fr][fc] = __builtin_amdgcn_wmma_f32_16x16x32_bf16(
                    false, ahi[fr], false, blo, (short)0, acc[fr][fc], false, false);
                acc[fr][fc] = __builtin_amdgcn_wmma_f32_16x16x32_bf16(
                    false, alo[fr], false, bhi, (short)0, acc[fr][fc], false, false);
            }
        }
    }

    // Epilogue: C/D layout: VGPR g, lane -> local row = 8h+g (within 16), col = l15
    float rmin[16], cmin[4];
    #pragma unroll
    for (int i = 0; i < 16; ++i) rmin[i] = __uint_as_float(INF_BITS);
    #pragma unroll
    for (int i = 0; i < 4; ++i)  cmin[i] = __uint_as_float(INF_BITS);
    float tsum = 0.f;

    #pragma unroll
    for (int fr = 0; fr < 2; ++fr) {
        #pragma unroll
        for (int fc = 0; fc < 4; ++fc) {
            const int colLocal = 64 * wc + 16 * fc + l15;
            const float nb = normB[tileN + colLocal];
            #pragma unroll
            for (int g = 0; g < 8; ++g) {
                const int rowLocal = 32 * wr + 16 * fr + 8 * h + g;
                const float na = normA[tileM + rowLocal];
                const float dot = acc[fr][fc][g];
                const float sq = na + nb - 2.0f * dot;
                const float dd = sqrtf(fmaxf(sq, 0.0f));
                tsum += dd;
                rmin[fr * 8 + g] = fminf(rmin[fr * 8 + g], dd);
                cmin[fc]         = fminf(cmin[fc], dd);
            }
        }
    }

    // LDS min reduction (order-independent -> deterministic)
    #pragma unroll
    for (int fc = 0; fc < 4; ++fc)
        atomicMin(&lds_min_c[64 * wc + 16 * fc + l15], __float_as_uint(cmin[fc]));
    #pragma unroll
    for (int fr = 0; fr < 2; ++fr)
        #pragma unroll
        for (int g = 0; g < 8; ++g)
            atomicMin(&lds_min_r[32 * wr + 16 * fr + 8 * h + g],
                      __float_as_uint(rmin[fr * 8 + g]));

    // Deterministic tree sum of the tile
    ssum[tid] = tsum;
    __syncthreads();
    #pragma unroll
    for (int s = 128; s > 0; s >>= 1) {
        if (tid < s) ssum[tid] += ssum[tid + s];
        __syncthreads();
    }
    if (tid == 0) blockSums[blockIdx.y * gridDim.x + blockIdx.x] = ssum[0];

    if (tid < 128) {
        atomicMin(&posR[tileM + tid], lds_min_r[tid]);
        atomicMin(&posC[tileN + tid], lds_min_c[tid]);
    }
}

// ---------------------------------------------------------------------------
// Kernel 3: final combine (single block, fixed-order double reductions).
// loss = 0.5*(2*margin - mean(posR) - mean(posC) + 2*Sd/(N*M) - margin/M - margin/N)
// ---------------------------------------------------------------------------
__global__ void finalize_kernel(const float* __restrict__ blockSums, int nBlocks,
                                const unsigned int* __restrict__ posR,
                                const unsigned int* __restrict__ posC,
                                float* __restrict__ out) {
    __shared__ double sd[256];
    const int tid = threadIdx.x;

    // --- Sd ---
    double v = 0.0;
    for (int i = tid; i < nBlocks; i += 256) v += (double)blockSums[i];
    sd[tid] = v; __syncthreads();
    for (int s = 128; s > 0; s >>= 1) { if (tid < s) sd[tid] += sd[tid + s]; __syncthreads(); }
    double Sd = sd[0]; __syncthreads();

    // --- sum posR ---
    v = 0.0;
    for (int i = tid; i < NROWS; i += 256) v += (double)__uint_as_float(posR[i]);
    sd[tid] = v; __syncthreads();
    for (int s = 128; s > 0; s >>= 1) { if (tid < s) sd[tid] += sd[tid + s]; __syncthreads(); }
    double sumR = sd[0]; __syncthreads();

    // --- sum posC ---
    v = 0.0;
    for (int i = tid; i < NROWS; i += 256) v += (double)__uint_as_float(posC[i]);
    sd[tid] = v; __syncthreads();
    for (int s = 128; s > 0; s >>= 1) { if (tid < s) sd[tid] += sd[tid + s]; __syncthreads(); }
    double sumC = sd[0];

    if (tid == 0) {
        const double N = (double)NROWS, M = (double)NROWS;
        double meanR = sumR / N;
        double meanC = sumC / M;
        double loss = 0.5 * (2.0 * MARGIN - meanR - meanC
                             + 2.0 * Sd / (N * M) - MARGIN / M - MARGIN / N);
        out[0] = (float)loss;
    }
}

// ---------------------------------------------------------------------------
extern "C" void kernel_launch(void* const* d_in, const int* in_sizes, int n_in,
                              void* d_out, int out_size, void* d_ws, size_t ws_size,
                              hipStream_t stream) {
    const float* A = (const float*)d_in[0];
    const float* B = (const float*)d_in[1];
    char* ws = (char*)d_ws;

    unsigned short* Ahi = (unsigned short*)(ws + WS_AHI);
    unsigned short* Alo = (unsigned short*)(ws + WS_ALO);
    unsigned short* Bhi = (unsigned short*)(ws + WS_BHI);
    unsigned short* Blo = (unsigned short*)(ws + WS_BLO);
    float*        normA = (float*)(ws + WS_NA);
    float*        normB = (float*)(ws + WS_NB);
    unsigned int*  posR = (unsigned int*)(ws + WS_POSR);
    unsigned int*  posC = (unsigned int*)(ws + WS_POSC);
    float*    blockSums = (float*)(ws + WS_BSUM);

    init_mins_kernel<<<(2 * NROWS + 255) / 256, 256, 0, stream>>>(posR, posC);
    prep_kernel<<<dim3(NROWS, 2), DDIM, 0, stream>>>(A, B, Ahi, Alo, Bhi, Blo, normA, normB);

    dim3 grid(NROWS / 128, NROWS / 128);  // 64 x 64 tiles
    dist_tile_kernel<<<grid, 256, 0, stream>>>(Ahi, Alo, Bhi, Blo, normA, normB,
                                               posR, posC, blockSums);

    finalize_kernel<<<1, 256, 0, stream>>>(blockSums, (int)(grid.x * grid.y),
                                           posR, posC, (float*)d_out);
}
